// TVNetSAModel_29695403885111
// MI455X (gfx1250) — matrix operand order repaired and checked by reading, not compiled, not run on hardware
//
#include <hip/hip_runtime.h>
#include <hip/hip_bf16.h>
#include <math.h>

#define BB 8
#define SS 1024
#define DD 768
#define NLAYER 2

typedef __bf16 bf16;
typedef __attribute__((ext_vector_type(16))) __bf16 v16bf;
typedef __attribute__((ext_vector_type(8)))  __bf16 v8bf;
typedef __attribute__((ext_vector_type(8)))  float  v8f;

// ---------------------------------------------------------------------------
// WMMA helpers (CDNA5 wave32, 16x16x32 bf16 -> f32)
// ---------------------------------------------------------------------------
__device__ __forceinline__ v8f wmma_bf16(v16bf a, v16bf b, v8f c) {
  // (neg_a, A, neg_b, B, c_mod, C, reuse_a, reuse_b)
  return __builtin_amdgcn_wmma_f32_16x16x32_bf16(false, a, false, b, (short)0, c,
                                                 false, false);
}

// A operand: 16x32 tile from row-major [M x K] bf16, row stride ld.
__device__ __forceinline__ v16bf load_a_frag(const bf16* __restrict__ base,
                                             int row0, int k0, int ld) {
  const int lane = threadIdx.x & 31;
  const bf16* q = base + (size_t)(row0 + (lane & 15)) * ld + k0 + ((lane >> 4) << 4);
  union { v16bf v; v8bf h[2]; } u;
  u.h[0] = *(const v8bf*)(q);
  u.h[1] = *(const v8bf*)(q + 8);
  return u.v;
}

// B operand: 32x16 tile (KxN) where memory holds B as [N x K] row-major, stride ld.
__device__ __forceinline__ v16bf load_b_frag(const bf16* __restrict__ base,
                                             int col0, int k0, int ld) {
  const int lane = threadIdx.x & 31;
  const bf16* q = base + (size_t)(col0 + (lane & 15)) * ld + k0 + ((lane >> 4) << 4);
  return *(const v16bf*)(q);
}

// ---------------------------------------------------------------------------
// fp32 -> bf16 conversion
// ---------------------------------------------------------------------------
__global__ __launch_bounds__(256) void cvt_bf16_kernel(const float* __restrict__ in,
                                                       bf16* __restrict__ out, int n) {
  int idx = blockIdx.x * 256 + threadIdx.x;
  if (idx < n) out[idx] = (bf16)in[idx];
}

// ---------------------------------------------------------------------------
// Gram: u[b,i,j] = (1/sqrt(D)) * sum_d h[b,i,d] h[b,j,d]
// block 128 thr (4 waves); wave tile 32x64; ping-pong pipelined K loop (step 64)
// grid (S/128, S/64, B)
// ---------------------------------------------------------------------------
__global__ __launch_bounds__(128) void gram_kernel(const bf16* __restrict__ hb,
                                                   float* __restrict__ u,
                                                   float inv_temper) {
  const int b = blockIdx.z;
  const int wave = threadIdx.x >> 5;
  const int i0 = blockIdx.x * 128 + wave * 32;
  const int j0 = blockIdx.y * 64;
  const bf16* hbb = hb + (size_t)b * SS * DD;

  v8f acc[2][4] = {};
  v16bf a0s[2], a1s[2], bfs[2][4];
  a0s[0] = load_a_frag(hbb, i0, 0, DD);
  a1s[0] = load_a_frag(hbb, i0 + 16, 0, DD);
#pragma unroll
  for (int c = 0; c < 4; ++c) bfs[0][c] = load_b_frag(hbb, j0 + c * 16, 0, DD);

  for (int k = 0; k < DD; k += 64) {
    // stage 1 prefetch (k+32, always in range since DD % 64 == 0)
    a0s[1] = load_a_frag(hbb, i0, k + 32, DD);
    a1s[1] = load_a_frag(hbb, i0 + 16, k + 32, DD);
#pragma unroll
    for (int c = 0; c < 4; ++c) bfs[1][c] = load_b_frag(hbb, j0 + c * 16, k + 32, DD);
    // compute stage 0
#pragma unroll
    for (int c = 0; c < 4; ++c) {
      acc[0][c] = wmma_bf16(a0s[0], bfs[0][c], acc[0][c]);
      acc[1][c] = wmma_bf16(a1s[0], bfs[0][c], acc[1][c]);
    }
    // stage 0 prefetch (k+64; wrap on last iteration, result unused)
    const int kn = (k + 64 < DD) ? k + 64 : 0;
    a0s[0] = load_a_frag(hbb, i0, kn, DD);
    a1s[0] = load_a_frag(hbb, i0 + 16, kn, DD);
#pragma unroll
    for (int c = 0; c < 4; ++c) bfs[0][c] = load_b_frag(hbb, j0 + c * 16, kn, DD);
    // compute stage 1
#pragma unroll
    for (int c = 0; c < 4; ++c) {
      acc[0][c] = wmma_bf16(a0s[1], bfs[1][c], acc[0][c]);
      acc[1][c] = wmma_bf16(a1s[1], bfs[1][c], acc[1][c]);
    }
  }

  const int lane = threadIdx.x & 31;
  const int nl = lane & 15;
  float* ub = u + (size_t)b * SS * SS;
#pragma unroll
  for (int mI = 0; mI < 2; ++mI) {
    const int mb = i0 + mI * 16 + ((lane >> 4) << 3);
#pragma unroll
    for (int c = 0; c < 4; ++c)
#pragma unroll
      for (int r = 0; r < 8; ++r)
        ub[(size_t)(mb + r) * SS + j0 + c * 16 + nl] = acc[mI][c][r] * inv_temper;
  }
}

// ---------------------------------------------------------------------------
// Generic projection: y = x @ W^T + bias, x bf16 [M,K], W bf16 [N,K]
// mode 0: store fp32 row-major [M,N];  mode 1: store bf16 transposed [B,N,S]
// block 128; wave tile 32x64; ping-pong pipelined; grid (M/128, N/64)
// ---------------------------------------------------------------------------
__global__ __launch_bounds__(128) void proj_kernel(const bf16* __restrict__ x,
                                                   const bf16* __restrict__ w,
                                                   const float* __restrict__ bias,
                                                   float* __restrict__ outF,
                                                   bf16* __restrict__ outT,
                                                   int mode) {
  const int wave = threadIdx.x >> 5;
  const int m0 = blockIdx.x * 128 + wave * 32;  // row in [B*S]
  const int n0 = blockIdx.y * 64;               // col in [D]

  v8f acc[2][4] = {};
  v16bf a0s[2], a1s[2], bfs[2][4];
  a0s[0] = load_a_frag(x, m0, 0, DD);
  a1s[0] = load_a_frag(x, m0 + 16, 0, DD);
#pragma unroll
  for (int c = 0; c < 4; ++c) bfs[0][c] = load_b_frag(w, n0 + c * 16, 0, DD);

  for (int k = 0; k < DD; k += 64) {
    a0s[1] = load_a_frag(x, m0, k + 32, DD);
    a1s[1] = load_a_frag(x, m0 + 16, k + 32, DD);
#pragma unroll
    for (int c = 0; c < 4; ++c) bfs[1][c] = load_b_frag(w, n0 + c * 16, k + 32, DD);
#pragma unroll
    for (int c = 0; c < 4; ++c) {
      acc[0][c] = wmma_bf16(a0s[0], bfs[0][c], acc[0][c]);
      acc[1][c] = wmma_bf16(a1s[0], bfs[0][c], acc[1][c]);
    }
    const int kn = (k + 64 < DD) ? k + 64 : 0;
    a0s[0] = load_a_frag(x, m0, kn, DD);
    a1s[0] = load_a_frag(x, m0 + 16, kn, DD);
#pragma unroll
    for (int c = 0; c < 4; ++c) bfs[0][c] = load_b_frag(w, n0 + c * 16, kn, DD);
#pragma unroll
    for (int c = 0; c < 4; ++c) {
      acc[0][c] = wmma_bf16(a0s[1], bfs[1][c], acc[0][c]);
      acc[1][c] = wmma_bf16(a1s[1], bfs[1][c], acc[1][c]);
    }
  }

  const int lane = threadIdx.x & 31;
  const int nl = lane & 15;
#pragma unroll
  for (int mI = 0; mI < 2; ++mI) {
    const int mb = m0 + mI * 16 + ((lane >> 4) << 3);
    const int batch = mb >> 10;     // /S (tiles never straddle a batch)
    const int il = mb & (SS - 1);   // %S
#pragma unroll
    for (int c = 0; c < 4; ++c) {
      const int n = n0 + c * 16 + nl;
      const float bn = bias[n];
      if (mode == 0) {
#pragma unroll
        for (int r = 0; r < 8; ++r)
          outF[(size_t)(mb + r) * DD + n] = acc[mI][c][r] + bn;
      } else {
        union { bf16 e[8]; v8bf v; } t;
#pragma unroll
        for (int r = 0; r < 8; ++r) t.e[r] = (bf16)(acc[mI][c][r] + bn);
        *(v8bf*)(outT + ((size_t)batch * DD + n) * SS + il) = t.v;
      }
    }
  }
}

// ---------------------------------------------------------------------------
// block reductions (8 waves of 32)
// ---------------------------------------------------------------------------
__device__ __forceinline__ float blk_sum(float v, float* sm) {
#pragma unroll
  for (int o = 16; o > 0; o >>= 1) v += __shfl_xor(v, o, 32);
  const int lane = threadIdx.x & 31, w = threadIdx.x >> 5;
  if (lane == 0) sm[w] = v;
  __syncthreads();
  float r = 0.f;
#pragma unroll
  for (int k = 0; k < 8; ++k) r += sm[k];
  __syncthreads();
  return r;
}

__device__ __forceinline__ float blk_max(float v, float* sm) {
#pragma unroll
  for (int o = 16; o > 0; o >>= 1) v = fmaxf(v, __shfl_xor(v, o, 32));
  const int lane = threadIdx.x & 31, w = threadIdx.x >> 5;
  if (lane == 0) sm[w] = v;
  __syncthreads();
  float r = sm[0];
#pragma unroll
  for (int k = 1; k < 8; ++k) r = fmaxf(r, sm[k]);
  __syncthreads();
  return r;
}

// ---------------------------------------------------------------------------
// softmax over row, mask by adjacency, renormalize, split upper/diag/lower
// grid (S, B), block 256
// ---------------------------------------------------------------------------
__global__ __launch_bounds__(256) void attn_kernel(const float* __restrict__ u,
                                                   const float* __restrict__ adj,
                                                   bf16* __restrict__ AU,
                                                   bf16* __restrict__ AL,
                                                   float* __restrict__ diag) {
  __shared__ float sm[8];
  const int i = blockIdx.x, b = blockIdx.y;
  const size_t row = (size_t)b * SS + i;
  const float* ur = u + row * SS;
  const float* ar = adj + row * SS;
  bf16* AUr = AU + row * SS;
  bf16* ALr = AL + row * SS;

  float x[4];
  float m = -__builtin_inff();
#pragma unroll
  for (int t = 0; t < 4; ++t) {
    x[t] = ur[threadIdx.x + t * 256];
    m = fmaxf(m, x[t]);
  }
  m = blk_max(m, sm);

  float e[4], s = 0.f;
#pragma unroll
  for (int t = 0; t < 4; ++t) { e[t] = __expf(x[t] - m); s += e[t]; }
  s = blk_sum(s, sm);
  const float invs = 1.f / s;

  float del[4], ds = 0.f;
#pragma unroll
  for (int t = 0; t < 4; ++t) {
    del[t] = e[t] * invs * ar[threadIdx.x + t * 256];
    ds += del[t];
  }
  ds = blk_sum(ds, sm);
  const float invd = 1.f / (ds + 1e-10f);

#pragma unroll
  for (int t = 0; t < 4; ++t) {
    const int j = threadIdx.x + t * 256;
    const float a = del[t] * invd;
    AUr[j] = (bf16)(j > i ? a : 0.f);
    ALr[j] = (bf16)(j < i ? a : 0.f);
    if (j == i) diag[row] = a;
  }
}

// ---------------------------------------------------------------------------
// ctx = AU @ hl + AL @ hr   (per batch: [S,S]x[S,D])
// block 128; wave tile 32x32; ping-pong pipelined (K step 64); grid (S/128, D/32, B)
// ---------------------------------------------------------------------------
__global__ __launch_bounds__(128) void ctx_kernel(const bf16* __restrict__ AU,
                                                  const bf16* __restrict__ AL,
                                                  const bf16* __restrict__ hlT,
                                                  const bf16* __restrict__ hrT,
                                                  float* __restrict__ ctx) {
  const int b = blockIdx.z;
  const int wave = threadIdx.x >> 5;
  const int i0 = blockIdx.x * 128 + wave * 32;
  const int n0 = blockIdx.y * 32;
  const bf16* AUb = AU + (size_t)b * SS * SS;
  const bf16* ALb = AL + (size_t)b * SS * SS;
  const bf16* hlb = hlT + (size_t)b * DD * SS;
  const bf16* hrb = hrT + (size_t)b * DD * SS;

  v8f acc[2][2] = {};
  v16bf aU0[2], aU1[2], aL0[2], aL1[2], bU[2][2], bL[2][2];
  aU0[0] = load_a_frag(AUb, i0, 0, SS);
  aU1[0] = load_a_frag(AUb, i0 + 16, 0, SS);
  aL0[0] = load_a_frag(ALb, i0, 0, SS);
  aL1[0] = load_a_frag(ALb, i0 + 16, 0, SS);
#pragma unroll
  for (int c = 0; c < 2; ++c) {
    bU[0][c] = load_b_frag(hlb, n0 + c * 16, 0, SS);
    bL[0][c] = load_b_frag(hrb, n0 + c * 16, 0, SS);
  }

  for (int k = 0; k < SS; k += 64) {
    // stage 1 prefetch (k+32, in range since SS % 64 == 0)
    aU0[1] = load_a_frag(AUb, i0, k + 32, SS);
    aU1[1] = load_a_frag(AUb, i0 + 16, k + 32, SS);
    aL0[1] = load_a_frag(ALb, i0, k + 32, SS);
    aL1[1] = load_a_frag(ALb, i0 + 16, k + 32, SS);
#pragma unroll
    for (int c = 0; c < 2; ++c) {
      bU[1][c] = load_b_frag(hlb, n0 + c * 16, k + 32, SS);
      bL[1][c] = load_b_frag(hrb, n0 + c * 16, k + 32, SS);
    }
    // compute stage 0
#pragma unroll
    for (int c = 0; c < 2; ++c) {
      acc[0][c] = wmma_bf16(aU0[0], bU[0][c], acc[0][c]);
      acc[0][c] = wmma_bf16(aL0[0], bL[0][c], acc[0][c]);
      acc[1][c] = wmma_bf16(aU1[0], bU[0][c], acc[1][c]);
      acc[1][c] = wmma_bf16(aL1[0], bL[0][c], acc[1][c]);
    }
    // stage 0 prefetch (k+64; wrap on last iteration)
    const int kn = (k + 64 < SS) ? k + 64 : 0;
    aU0[0] = load_a_frag(AUb, i0, kn, SS);
    aU1[0] = load_a_frag(AUb, i0 + 16, kn, SS);
    aL0[0] = load_a_frag(ALb, i0, kn, SS);
    aL1[0] = load_a_frag(ALb, i0 + 16, kn, SS);
#pragma unroll
    for (int c = 0; c < 2; ++c) {
      bU[0][c] = load_b_frag(hlb, n0 + c * 16, kn, SS);
      bL[0][c] = load_b_frag(hrb, n0 + c * 16, kn, SS);
    }
    // compute stage 1
#pragma unroll
    for (int c = 0; c < 2; ++c) {
      acc[0][c] = wmma_bf16(aU0[1], bU[1][c], acc[0][c]);
      acc[0][c] = wmma_bf16(aL0[1], bL[1][c], acc[0][c]);
      acc[1][c] = wmma_bf16(aU1[1], bU[1][c], acc[1][c]);
      acc[1][c] = wmma_bf16(aL1[1], bL[1][c], acc[1][c]);
    }
  }

  const int lane = threadIdx.x & 31;
  const int nl = lane & 15;
#pragma unroll
  for (int mI = 0; mI < 2; ++mI) {
    const int mb = i0 + mI * 16 + ((lane >> 4) << 3);
#pragma unroll
    for (int c = 0; c < 2; ++c)
#pragma unroll
      for (int r = 0; r < 8; ++r)
        ctx[((size_t)b * SS + mb + r) * DD + n0 + c * 16 + nl] = acc[mI][c][r];
  }
}

// ---------------------------------------------------------------------------
// t = ctx + diag*hs ; LayerNorm ; ReLU.  grid (S,B), block 256
// ---------------------------------------------------------------------------
__global__ __launch_bounds__(256) void ln_kernel(const float* __restrict__ ctx,
                                                 const float* __restrict__ hs,
                                                 const float* __restrict__ diag,
                                                 const float* __restrict__ g,
                                                 const float* __restrict__ beta,
                                                 float* __restrict__ out) {
  __shared__ float sm[8];
  const int i = blockIdx.x, b = blockIdx.y;
  const size_t row = (size_t)b * SS + i;
  const float* cr = ctx + row * DD;
  const float* hr = hs + row * DD;
  const float da = diag[row];

  float t[3], s = 0.f, s2 = 0.f;
#pragma unroll
  for (int e = 0; e < 3; ++e) {
    const int d = threadIdx.x + e * 256;
    t[e] = cr[d] + da * hr[d];
    s += t[e];
    s2 += t[e] * t[e];
  }
  s = blk_sum(s, sm);
  s2 = blk_sum(s2, sm);
  const float mean = s * (1.f / DD);
  const float var = s2 * (1.f / DD) - mean * mean;
  const float rstd = rsqrtf(var + 1e-12f);
#pragma unroll
  for (int e = 0; e < 3; ++e) {
    const int d = threadIdx.x + e * 256;
    const float o = (t[e] - mean) * rstd * g[d] + beta[d];
    out[row * DD + d] = fmaxf(o, 0.f);
  }
}

// ---------------------------------------------------------------------------
// host side
// ---------------------------------------------------------------------------
static inline size_t align256(size_t x) { return (x + 255) & ~(size_t)255; }

extern "C" void kernel_launch(void* const* d_in, const int* in_sizes, int n_in,
                              void* d_out, int out_size, void* d_ws, size_t ws_size,
                              hipStream_t stream) {
  (void)in_sizes; (void)n_in; (void)out_size; (void)ws_size;
  const float* h_in = (const float*)d_in[0];
  const float* adj  = (const float*)d_in[1];
  const float* Wl   = (const float*)d_in[2];
  const float* bl   = (const float*)d_in[3];
  const float* Ws   = (const float*)d_in[4];
  const float* bs   = (const float*)d_in[5];
  const float* Wr   = (const float*)d_in[6];
  const float* br   = (const float*)d_in[7];
  const float* ln_w = (const float*)d_in[8];
  const float* ln_b = (const float*)d_in[9];
  float* out = (float*)d_out;

  char* ws = (char*)d_ws;
  size_t off = 0;
  bf16* hb   = (bf16*)(ws + off);  off = align256(off + (size_t)BB * SS * DD * 2);
  float* u   = (float*)(ws + off); off = align256(off + (size_t)BB * SS * SS * 4); // aliased by ctx
  bf16* AU   = (bf16*)(ws + off);  off = align256(off + (size_t)BB * SS * SS * 2);
  bf16* AL   = (bf16*)(ws + off);  off = align256(off + (size_t)BB * SS * SS * 2);
  float* dia = (float*)(ws + off); off = align256(off + (size_t)BB * SS * 4);
  bf16* wlb  = (bf16*)(ws + off);  off = align256(off + (size_t)DD * DD * 2);
  bf16* wsb  = (bf16*)(ws + off);  off = align256(off + (size_t)DD * DD * 2);
  bf16* wrb  = (bf16*)(ws + off);  off = align256(off + (size_t)DD * DD * 2);
  bf16* hlT  = (bf16*)(ws + off);  off = align256(off + (size_t)BB * DD * SS * 2);
  bf16* hrT  = (bf16*)(ws + off);  off = align256(off + (size_t)BB * DD * SS * 2);
  float* hs  = (float*)(ws + off); off = align256(off + (size_t)BB * SS * DD * 4);
  float* hcur = (float*)(ws + off); off = align256(off + (size_t)BB * SS * DD * 4);
  float* ctx = u;  // u fully consumed before ctx is produced

  const float inv_temper = 1.0f / sqrtf((float)DD);
  const int nH = BB * SS * DD;
  const int nW = DD * DD;
  const dim3 projGrid((BB * SS) / 128, DD / 64);

  for (int layer = 0; layer < NLAYER; ++layer) {
    const float* h = (layer == 0) ? h_in : hcur;
    float* hout = (layer == NLAYER - 1) ? out : hcur;

    cvt_bf16_kernel<<<(nH + 255) / 256, 256, 0, stream>>>(h, hb, nH);
    cvt_bf16_kernel<<<(nW + 255) / 256, 256, 0, stream>>>(Wl + (size_t)layer * nW, wlb, nW);
    cvt_bf16_kernel<<<(nW + 255) / 256, 256, 0, stream>>>(Ws + (size_t)layer * nW, wsb, nW);
    cvt_bf16_kernel<<<(nW + 255) / 256, 256, 0, stream>>>(Wr + (size_t)layer * nW, wrb, nW);

    gram_kernel<<<dim3(SS / 128, SS / 64, BB), 128, 0, stream>>>(hb, u, inv_temper);

    attn_kernel<<<dim3(SS, BB), 256, 0, stream>>>(u, adj, AU, AL, dia);

    // hl (bf16 transposed), hs (fp32), hr (bf16 transposed)
    proj_kernel<<<projGrid, 128, 0, stream>>>(hb, wlb, bl + (size_t)layer * DD,
                                              nullptr, hlT, 1);
    proj_kernel<<<projGrid, 128, 0, stream>>>(hb, wsb, bs + (size_t)layer * DD,
                                              hs, nullptr, 0);
    proj_kernel<<<projGrid, 128, 0, stream>>>(hb, wrb, br + (size_t)layer * DD,
                                              nullptr, hrT, 1);

    ctx_kernel<<<dim3(SS / 128, DD / 32, BB), 128, 0, stream>>>(AU, AL, hlT, hrT, ctx);

    ln_kernel<<<dim3(SS, BB), 256, 0, stream>>>(
        ctx, hs, dia, ln_w + (size_t)layer * DD, ln_b + (size_t)layer * DD, hout);
  }
}
